// TopkMoE_50946902065585
// MI455X (gfx1250) — matrix operand order repaired and checked by reading, not compiled, not run on hardware
//
#include <hip/hip_runtime.h>
#include <hip/hip_bf16.h>

// Problem constants (reference: B=4, S=2048, D=1024, E=8, k=2)
#define NTOK    8192
#define DIM     1024
#define NEXP    8
#define TILE_M  32                  // tokens per block / per wave (2 WMMA M-tiles)
#define NTILES  (NTOK / TILE_M)     // 256 worst-case token tiles per expert

typedef __attribute__((ext_vector_type(16))) __bf16 v16bf;
typedef __attribute__((ext_vector_type(8)))  float  v8f;

__device__ __forceinline__ unsigned short bfbits(float f) {
    // round-to-nearest-even f32 -> bf16 raw bits
    unsigned u = __builtin_bit_cast(unsigned, f);
    u += 0x7FFFu + ((u >> 16) & 1u);
    return (unsigned short)(u >> 16);
}
__device__ __forceinline__ unsigned pack2(float lo, float hi) {
    return ((unsigned)bfbits(hi) << 16) | (unsigned)bfbits(lo);
}

// Fragment = 16 bf16 (32 B) loaded as two 16-B chunks
union Frag { v16bf v; uint4 u[2]; };

// ---------------------------------------------------------------------------
// Kernel 0: zero the per-expert token counters
// ---------------------------------------------------------------------------
__global__ void moe_init(int* __restrict__ counts) {
    if (threadIdx.x < NEXP) counts[threadIdx.x] = 0;
}

// ---------------------------------------------------------------------------
// Kernel 1: streaming f32 -> bf16 conversion, 8 elements / thread.
// ---------------------------------------------------------------------------
__global__ void __launch_bounds__(256)
f32_to_bf16(const float* __restrict__ src, unsigned int* __restrict__ dst) {
    const int i = blockIdx.x * 256 + threadIdx.x;   // handles elems [8i, 8i+8)
    const float4* s = (const float4*)src + (size_t)i * 2;
    float4 f0 = s[0], f1 = s[1];
    uint4 o;
    o.x = pack2(f0.x, f0.y);
    o.y = pack2(f0.z, f0.w);
    o.z = pack2(f1.x, f1.y);
    o.w = pack2(f1.z, f1.w);
    ((uint4*)dst)[i] = o;
}

// ---------------------------------------------------------------------------
// Kernel 2: router. One wave32 per token: 8 dot products (D=1024), softmax,
// top-2 -> winning expert = largest index among top-2 probs (reference's
// overwrite semantics); compact token id into per-expert list, store gate.
// ---------------------------------------------------------------------------
__global__ void moe_router(const float* __restrict__ x,
                           const float* __restrict__ Wr,
                           const float* __restrict__ br,
                           float* __restrict__ wtok,
                           int*  __restrict__ counts,
                           int*  __restrict__ list) {
    const int lane = threadIdx.x & 31;
    const int wave = threadIdx.x >> 5;
    const int t    = blockIdx.x * 8 + wave;   // grid sized exactly: NTOK/8 blocks
    const float* xr = x + (size_t)t * DIM;

    float acc[NEXP];
#pragma unroll
    for (int e = 0; e < NEXP; ++e) acc[e] = 0.0f;

    for (int d = lane; d < DIM; d += 32) {
        float xv = xr[d];
#pragma unroll
        for (int e = 0; e < NEXP; ++e)
            acc[e] = __builtin_fmaf(xv, Wr[e * DIM + d], acc[e]);
    }
#pragma unroll
    for (int e = 0; e < NEXP; ++e) {
        float v = acc[e];
#pragma unroll
        for (int off = 16; off > 0; off >>= 1) v += __shfl_xor(v, off, 32);
        acc[e] = v;
    }

    if (lane == 0) {
        float lg[NEXP], mx = -3.0e38f;
#pragma unroll
        for (int e = 0; e < NEXP; ++e) { lg[e] = acc[e] + br[e]; mx = fmaxf(mx, lg[e]); }
        float s = 0.0f;
#pragma unroll
        for (int e = 0; e < NEXP; ++e) { lg[e] = __expf(lg[e] - mx); s += lg[e]; }
        int i1 = 0;
#pragma unroll
        for (int e = 1; e < NEXP; ++e) if (lg[e] > lg[i1]) i1 = e;
        int i2 = (i1 == 0) ? 1 : 0;
#pragma unroll
        for (int e = 0; e < NEXP; ++e) if (e != i1 && lg[e] > lg[i2]) i2 = e;
        const int imax = (i1 > i2) ? i1 : i2;
        wtok[t] = lg[imax] / s;
        int pos = atomicAdd(&counts[imax], 1);
        list[imax * NTOK + pos] = t;
    }
}

// ---------------------------------------------------------------------------
// Kernel 3: grouped expert GEMM, pre-converted bf16 operands, f32 accumulate.
// grid.x = NEXP * NTILES (expert, 32-token tile), grid.y = DIM/256.
// Block = 256 threads = 8 waves. Wave w computes a 32x32 output tile as a
// 2x2 grid of 16x16 WMMA accumulators (c00,c01,c10,c11) at feature offset
// blockIdx.y*256 + w*32, looping K=DIM in steps of 32.
// Hot loop: 8 x global_load_b128 + 4 x v_wmma_f32_16x16x32_bf16 per K-step;
// 4 accumulators -> WMMA reuse distance 4 -> no hazard NOPs.
// ---------------------------------------------------------------------------
__global__ void __launch_bounds__(256)
moe_expert_gemm(const unsigned short* __restrict__ xb,
                const unsigned short* __restrict__ web,
                const float* __restrict__ be,
                const float* __restrict__ wtok,
                const int*  __restrict__ counts,
                const int*  __restrict__ list,
                float* __restrict__ out) {
    const int e      = blockIdx.x >> 8;            // / NTILES (256)
    const int tile   = blockIdx.x & (NTILES - 1);
    const int cnt    = counts[e];
    const int tstart = tile * TILE_M;
    if (tstart >= cnt) return;                     // uniform: EXEC all-1s for WMMA

    const int lane = threadIdx.x & 31;
    const int wave = threadIdx.x >> 5;
    const int n    = lane & 15;
    const int half = lane >> 4;
    const int f0   = blockIdx.y * 256 + wave * 32;

    const int* tl = list + e * NTOK + tstart;

    // A rows for this lane (m = n and m = n + 16), guarded gather
    const int tokA0 = (tstart + n      < cnt) ? tl[n]      : tl[0];
    const int tokA1 = (tstart + 16 + n < cnt) ? tl[16 + n] : tl[0];
    const unsigned short* rowA0 = xb + (size_t)tokA0 * DIM;
    const unsigned short* rowA1 = xb + (size_t)tokA1 * DIM;
    // B rows: this lane carries output features f0+n and f0+16+n (We[e] rows)
    const unsigned short* webE  = web + (size_t)e * DIM * DIM;
    const unsigned short* rowB0 = webE + (size_t)(f0 + n) * DIM;
    const unsigned short* rowB1 = webE + (size_t)(f0 + 16 + n) * DIM;

    v8f c00 = {}, c01 = {}, c10 = {}, c11 = {};
#pragma unroll 2
    for (int k0 = 0; k0 < DIM; k0 += 32) {
        // A: 16x32 bf16 (ISA 7.12.2): VGPR0..3 K=half*8+0..7, VGPR4..7 K=16+half*8+0..7
        Frag a0, a1, b0, b1;
        a0.u[0] = *(const uint4*)(rowA0 + k0 + half * 8);
        a0.u[1] = *(const uint4*)(rowA0 + k0 + 16 + half * 8);
        a1.u[0] = *(const uint4*)(rowA1 + k0 + half * 8);
        a1.u[1] = *(const uint4*)(rowA1 + k0 + 16 + half * 8);
        // B: 32x16 bf16: same K order as A: VGPR0..3 K=half*8+0..7, VGPR4..7 K=16+half*8+0..7
        b0.u[0] = *(const uint4*)(rowB0 + k0 + half * 8);
        b0.u[1] = *(const uint4*)(rowB0 + k0 + 16 + half * 8);
        b1.u[0] = *(const uint4*)(rowB1 + k0 + half * 8);
        b1.u[1] = *(const uint4*)(rowB1 + k0 + 16 + half * 8);

        c00 = __builtin_amdgcn_wmma_f32_16x16x32_bf16(
                  false, a0.v, false, b0.v, (short)0, c00, false, false);
        c01 = __builtin_amdgcn_wmma_f32_16x16x32_bf16(
                  false, a0.v, false, b1.v, (short)0, c01, false, false);
        c10 = __builtin_amdgcn_wmma_f32_16x16x32_bf16(
                  false, a1.v, false, b0.v, (short)0, c10, false, false);
        c11 = __builtin_amdgcn_wmma_f32_16x16x32_bf16(
                  false, a1.v, false, b1.v, (short)0, c11, false, false);
    }

    // Epilogue: C layout: VGPR r, lanes 0-15 -> M=r, lanes 16-31 -> M=r+8
    const float bias0 = be[e * DIM + f0 + n];
    const float bias1 = be[e * DIM + f0 + 16 + n];
#pragma unroll
    for (int r = 0; r < 8; ++r) {
        const int M = half * 8 + r;
        if (tstart + M < cnt) {
            const int tok = tl[M];
            const float w = wtok[tok];
            float* o = out + (size_t)tok * DIM + f0 + n;
            o[0]  = (c00[r] + bias0) * w;
            o[16] = (c01[r] + bias1) * w;
        }
        if (tstart + 16 + M < cnt) {
            const int tok = tl[16 + M];
            const float w = wtok[tok];
            float* o = out + (size_t)tok * DIM + f0 + n;
            o[0]  = (c10[r] + bias0) * w;
            o[16] = (c11[r] + bias1) * w;
        }
    }
}

// ---------------------------------------------------------------------------
extern "C" void kernel_launch(void* const* d_in, const int* in_sizes, int n_in,
                              void* d_out, int out_size, void* d_ws, size_t ws_size,
                              hipStream_t stream) {
    const float* x  = (const float*)d_in[0];   // [4,2048,1024]
    const float* Wr = (const float*)d_in[1];   // [8,1024]
    const float* br = (const float*)d_in[2];   // [8]
    const float* We = (const float*)d_in[3];   // [8,1024,1024]
    const float* be = (const float*)d_in[4];   // [8,1024]
    // d_in[5] = k (==2); overwrite semantics make it implicit via imax.
    float* out = (float*)d_out;

    // Workspace layout:
    //   [counts: 8 int | pad64 | wtok: NTOK f32 | list: NEXP*NTOK int |
    //    pad256 | xb: NTOK*DIM bf16 (16 MB) | web: NEXP*DIM*DIM bf16 (16 MB)]
    char*  ws     = (char*)d_ws;
    int*   counts = (int*)ws;
    float* wtok   = (float*)(ws + 64);
    int*   list   = (int*)(ws + 64 + (size_t)NTOK * 4);
    size_t base   = 64 + (size_t)NTOK * 4 + (size_t)NEXP * NTOK * 4;
    base = (base + 255) & ~(size_t)255;
    unsigned short* xb  = (unsigned short*)(ws + base);
    unsigned short* web = xb + (size_t)NTOK * DIM;

    moe_init<<<1, 32, 0, stream>>>(counts);

    // One-time bf16 conversions (streaming, 8 elem/thread)
    f32_to_bf16<<<(NTOK * DIM) / (256 * 8), 256, 0, stream>>>(x,  (unsigned int*)xb);
    f32_to_bf16<<<(NEXP * DIM * DIM) / (256 * 8), 256, 0, stream>>>(We, (unsigned int*)web);

    moe_router<<<NTOK / 8, 256, 0, stream>>>(x, Wr, br, wtok, counts, list);

    dim3 grid(NEXP * NTILES, DIM / 256);
    moe_expert_gemm<<<grid, 256, 0, stream>>>(xb, web, be, wtok, counts, list, out);
}
